// NumberGraphEncoder_36532991820103
// MI455X (gfx1250) — compile-verified, matrix operand order checked by reading, not run on hardware
//
#include <hip/hip_runtime.h>
#include <stdint.h>

// ---------------------------------------------------------------------------
// MI455X (gfx1250) fused GAT-VAE encoder.
// Compute-bound (~242 GFLOP vs ~50 MB traffic) -> all dense matmuls through
// v_wmma_f32_16x16x32_bf16 (wave32, f32 accumulate). Global operands in
// addrspace(1) (global_load_b128), embedding/adjacency staged with
// GLOBAL_LOAD_ASYNC_TO_LDS (ASYNCcnt), A-fragments reused across 2-4 column
// tiles per wave (multiple accumulators) to raise wmma:load ratio.
// ---------------------------------------------------------------------------

typedef __attribute__((ext_vector_type(16))) __bf16 v16bf;
typedef __attribute__((ext_vector_type(8)))  float  v8f;

typedef __attribute__((address_space(1))) const __bf16* gcbfp;
typedef __attribute__((address_space(1))) const float*  gcfp;
typedef __attribute__((address_space(1))) const int*    gcip;
typedef __attribute__((address_space(1))) float*        gfp;

#define SPB    8            // samples per block
#define NODES  6
#define ROWS   (SPB*NODES)  // 48 rows = 3 WMMA row tiles
#define EMB    128
#define HID    256
#define PROJ   512
#define REPR   (NODES*HID)  // 1536
#define LD     264          // padded bf16 LDS stride (row = 528 B = 33*16, 16B aligned)
#define LDX    520          // padded stride for x[*,512]
#define NEG_SLOPE 0.2f

#define WMMA_BF16(a, b, c) __builtin_amdgcn_wmma_f32_16x16x32_bf16( \
    false, (a), false, (b), (short)0, (c), false, false)

// ---- CDNA5 async global->LDS copies (ASYNCcnt) -----------------------------
static __device__ __forceinline__ void async_b128(uint32_t lds_off, uint64_t gaddr) {
  asm volatile("global_load_async_to_lds_b128 %0, %1, off"
               :: "v"(lds_off), "v"(gaddr) : "memory");
}
static __device__ __forceinline__ void async_b32(uint32_t lds_off, uint64_t gaddr) {
  asm volatile("global_load_async_to_lds_b32 %0, %1, off"
               :: "v"(lds_off), "v"(gaddr) : "memory");
}
static __device__ __forceinline__ void wait_async0() {
  asm volatile("s_wait_asynccnt 0x0" ::: "memory");
}
static __device__ __forceinline__ uint32_t lds_addr(const void* p) {
  return (uint32_t)(uintptr_t)p;   // generic LDS addr: low 32 bits = LDS offset
}

// ---- WMMA fragment loaders (16-bit A 16x32 / B 32x16 lane layout, ISA 7.12.2)
// lanes 0-15: row (A) / col (B) = lane, K in {k+0..7} U {k+16..23}
// lanes 16-31: row/col = lane-16, K in {k+8..15} U {k+24..31}
static __device__ __forceinline__ v16bf frag_lds(const __bf16* blk, int ld, int lane) {
  const int half = lane >> 4;
  const int r    = lane & 15;
  const __bf16* p = blk + r * ld;
  const int k0 = half ? 8 : 0;
  const int k1 = half ? 24 : 16;
  v16bf v;
#pragma unroll
  for (int e = 0; e < 8; ++e) v[e] = p[k0 + e];
#pragma unroll
  for (int e = 0; e < 8; ++e) v[8 + e] = p[k1 + e];
  return v;
}

static __device__ __forceinline__ v16bf frag_gbl(gcbfp blk, int ld, int lane) {
  const int half = lane >> 4;
  const int r    = lane & 15;
  gcbfp p = blk + r * ld;
  const int k0 = half ? 8 : 0;
  const int k1 = half ? 24 : 16;
  v16bf v;
#pragma unroll
  for (int e = 0; e < 8; ++e) v[e] = p[k0 + e];
#pragma unroll
  for (int e = 0; e < 8; ++e) v[8 + e] = p[k1 + e];
  return v;
}

// A-fragment for the flattened graph representation: row = sample (0..SPB-1),
// K index kk maps to LDS activation sA[(s*6 + kk/256)*LD + kk%256].
static __device__ __forceinline__ v16bf frag_repr(const __bf16* sA, int kBlock, int lane) {
  const int half = lane >> 4;
  const int s    = lane & 15;
  v16bf v;
  if (s >= SPB) {
#pragma unroll
    for (int e = 0; e < 16; ++e) v[e] = (__bf16)0.0f;
    return v;
  }
  const int k0 = kBlock + (half ? 8 : 0);
  const int k1 = kBlock + (half ? 24 : 16);
#pragma unroll
  for (int e = 0; e < 8; ++e) {
    int kk = k0 + e;
    v[e] = sA[(s * NODES + (kk >> 8)) * LD + (kk & 255)];
  }
#pragma unroll
  for (int e = 0; e < 8; ++e) {
    int kk = k1 + e;
    v[8 + e] = sA[(s * NODES + (kk >> 8)) * LD + (kk & 255)];
  }
  return v;
}

// ---- GAT layer matmul: sWh[48 x 256] = sA[48 x KD] @ W[KD x 256] -----------
// Each wave owns columns {wave, wave+8} for each of the 3 row tiles, sharing
// one A fragment per K-step between two accumulators (2 wmma per A load).
template<int KD>
static __device__ __forceinline__ void layer_matmul(const __bf16* sA, __bf16* sWh,
                                                    gcbfp gWt, int wave, int lane) {
  const int half = lane >> 4, n = lane & 15;
#pragma unroll
  for (int rt = 0; rt < 3; ++rt) {
    const int nt0 = wave, nt1 = wave + 8;
    v8f acc0 = {}, acc1 = {};
#pragma unroll 2
    for (int k = 0; k < KD; k += 32) {
      v16bf a  = frag_lds(sA + rt * 16 * LD + k, LD, lane);
      v16bf b0 = frag_gbl(gWt + nt0 * 16 * KD + k, KD, lane);
      v16bf b1 = frag_gbl(gWt + nt1 * 16 * KD + k, KD, lane);
      acc0 = WMMA_BF16(a, b0, acc0);
      acc1 = WMMA_BF16(a, b1, acc1);
    }
#pragma unroll
    for (int r = 0; r < 8; ++r) {
      sWh[(rt * 16 + r + half * 8) * LD + nt0 * 16 + n] = (__bf16)acc0[r];
      sWh[(rt * 16 + r + half * 8) * LD + nt1 * 16 + n] = (__bf16)acc1[r];
    }
  }
}

// ---- prep: fp32 -> bf16, optionally transposed to [N,K] for B operands -----
__global__ void cvt_bf16_kernel(const float* __restrict__ src, __bf16* __restrict__ dst,
                                int R, int C, int doTranspose) {
  int i = blockIdx.x * blockDim.x + threadIdx.x;
  if (i >= R * C) return;
  gcfp s = (gcfp)(uintptr_t)src;
  int r = i / C, c = i % C;
  __bf16 v = (__bf16)s[i];
  if (doTranspose) ((__attribute__((address_space(1))) __bf16*)(uintptr_t)dst)[c * R + r] = v;
  else             ((__attribute__((address_space(1))) __bf16*)(uintptr_t)dst)[i] = v;
}

// ---------------------------------------------------------------------------
__global__ __launch_bounds__(256)
void gat_vae_fused_kernel(const int*    __restrict__ numbers,
                          const float*  __restrict__ adj,
                          const __bf16* __restrict__ embb,
                          const __bf16* __restrict__ Wt0,
                          const __bf16* __restrict__ Wt1,
                          const __bf16* __restrict__ Wt2,
                          const float*  __restrict__ a0,
                          const float*  __restrict__ a1,
                          const float*  __restrict__ a2,
                          const __bf16* __restrict__ P1t,
                          const float*  __restrict__ P1b,
                          const __bf16* __restrict__ P2t,
                          const float*  __restrict__ P2b,
                          float* __restrict__ out, int Bn)
{
  __shared__ __bf16 sA [ROWS * LD];   // activations (bf16)
  __shared__ __bf16 sWh[ROWS * LD];   // matmul output / x buffer (bf16)
  __shared__ float  sAttn[ROWS * 8];
  __shared__ float  sSrc[ROWS];
  __shared__ float  sDst[ROWS];
  __shared__ float  sAdj[SPB * NODES * NODES];

  const int tid   = threadIdx.x;
  const int lane  = tid & 31;
  const int wave  = tid >> 5;
  const int sBase = blockIdx.x * SPB;

  const gcip gnum = (gcip)(uintptr_t)numbers;

  // Warm L2/WGP$ with the weight streams (global_prefetch_b8).
  if (tid < 8) {
    __builtin_prefetch((const void*)(Wt1 + tid * 8192), 0, 0);
    __builtin_prefetch((const void*)(P1t + tid * 98304), 0, 0);
  }

  // ---- async-stage embeddings into sA[48 x 128]: 16B chunks, GV mode ----
  for (int w = tid; w < ROWS * 16; w += 256) {
    const int i  = w >> 4;
    const int c8 = (w & 15) * 8;
    const int gs = sBase + i / NODES;
    if (gs < Bn) {
      const int vidx = gnum[gs * NODES + (i % NODES)];
      async_b128(lds_addr(&sA[i * LD + c8]),
                 (uint64_t)(uintptr_t)(embb + vidx * EMB + c8));
    } else {
#pragma unroll
      for (int e = 0; e < 8; ++e) sA[i * LD + c8 + e] = (__bf16)0.0f;
    }
  }
  // ---- async-stage adjacency tile ----
  for (int w = tid; w < SPB * 36; w += 256) {
    const int g = sBase * 36 + w;
    if (g < Bn * 36) async_b32(lds_addr(&sAdj[w]), (uint64_t)(uintptr_t)(adj + g));
    else             sAdj[w] = 1.0f;
  }
  wait_async0();
  __syncthreads();

  const __bf16* Wts[3] = {Wt0, Wt1, Wt2};
  const float*  avs[3] = {a0, a1, a2};

  for (int layer = 0; layer < 3; ++layer) {
    // ---- Wh = h @ W (templated K so the k-loop can pipeline) ----
    const gcbfp gWt = (gcbfp)(uintptr_t)Wts[layer];
    if (layer == 0) layer_matmul<EMB>(sA, sWh, gWt, wave, lane);
    else            layer_matmul<HID>(sA, sWh, gWt, wave, lane);
    __syncthreads();

    // ---- a_src / a_dst per row ----
    const gcfp gav = (gcfp)(uintptr_t)avs[layer];
    if (tid < ROWS) {
      float s0 = 0.f, s1 = 0.f;
      for (int c = 0; c < HID; ++c) {
        float w = (float)sWh[tid * LD + c];
        s0 += w * gav[c];
        s1 += w * gav[HID + c];
      }
      sSrc[tid] = s0;
      sDst[tid] = s1;
    }
    __syncthreads();

    // ---- masked leaky-ReLU softmax over 6 neighbors ----
    if (tid < ROWS) {
      const int s = tid / NODES, ii = tid % NODES;
      float e[NODES];
      float m = -3.4e38f;
#pragma unroll
      for (int j = 0; j < NODES; ++j) {
        float v = sSrc[tid] + sDst[s * NODES + j];
        v = (v > 0.f) ? v : NEG_SLOPE * v;
        if (sAdj[s * 36 + ii * NODES + j] == 0.f) v = -1e9f;
        e[j] = v;
        m = fmaxf(m, v);
      }
      float sum = 0.f;
#pragma unroll
      for (int j = 0; j < NODES; ++j) { e[j] = __expf(e[j] - m); sum += e[j]; }
      const float inv = 1.f / sum;
#pragma unroll
      for (int j = 0; j < NODES; ++j) sAttn[tid * 8 + j] = e[j] * inv;
    }
    __syncthreads();

    // ---- h = relu(attn @ Wh) -> sA ----
    for (int w = tid; w < ROWS * HID; w += 256) {
      const int i = w >> 8, c = w & 255;
      const int s = i / NODES;
      float acc = 0.f;
#pragma unroll
      for (int j = 0; j < NODES; ++j)
        acc += sAttn[i * 8 + j] * (float)sWh[(s * NODES + j) * LD + c];
      sA[i * LD + c] = (__bf16)fmaxf(acc, 0.f);
    }
    __syncthreads();
  }

  // ---- P1: x = relu(repr @ P1 + b), K=1536 ----
  // One shared A fragment (frag_repr) drives 4 accumulators per wave:
  // columns {wave, wave+8, wave+16, wave+24} -> 4 wmma per A load.
  __bf16* xbuf = sWh; // reuse as [16 x 512] with stride LDX
  {
    const gcbfp gP1 = (gcbfp)(uintptr_t)P1t;
    const gcfp  gB1 = (gcfp)(uintptr_t)P1b;
    v8f acc[4] = {{}, {}, {}, {}};
#pragma unroll 2
    for (int k = 0; k < REPR; k += 32) {
      v16bf a = frag_repr(sA, k, lane);
#pragma unroll
      for (int j = 0; j < 4; ++j) {
        v16bf b = frag_gbl(gP1 + (wave + 8 * j) * 16 * REPR + k, REPR, lane);
        acc[j] = WMMA_BF16(a, b, acc[j]);
      }
    }
    const int half = lane >> 4, n = lane & 15;
#pragma unroll
    for (int j = 0; j < 4; ++j) {
      const int nt = wave + 8 * j;
      const float bias = gB1[nt * 16 + n];
#pragma unroll
      for (int r = 0; r < 8; ++r)
        xbuf[(r + half * 8) * LDX + nt * 16 + n] = (__bf16)fmaxf(acc[j][r] + bias, 0.f);
    }
  }
  __syncthreads();

  // ---- P2: latent = x @ P2 + b -> (mu | logvar), 8 col tiles, K=512 ----
  {
    const gcbfp gP2 = (gcbfp)(uintptr_t)P2t;
    const gcfp  gB2 = (gcfp)(uintptr_t)P2b;
    const gfp   gout = (gfp)(uintptr_t)out;
    const int nt = wave;
    v8f acc = {};
#pragma unroll 2
    for (int k = 0; k < PROJ; k += 32) {
      v16bf afr = frag_lds(xbuf + k, LDX, lane);
      v16bf bfr = frag_gbl(gP2 + nt * 16 * PROJ + k, PROJ, lane);
      acc = WMMA_BF16(afr, bfr, acc);
    }
    const int half = lane >> 4, n = lane & 15;
    const int col  = nt * 16 + n;
    const float bias = gB2[col];
#pragma unroll
    for (int r = 0; r < 8; ++r) {
      const int s = r + half * 8;
      const int gs = sBase + s;
      if (s < SPB && gs < Bn) {
        const float v = acc[r] + bias;
        if (col < 64) gout[(size_t)gs * 64 + col] = v;                       // mu
        else gout[(size_t)Bn * 64 + (size_t)gs * 64 + (col - 64)] = v;       // logvar
      }
    }
  }
}

// ---------------------------------------------------------------------------
extern "C" void kernel_launch(void* const* d_in, const int* in_sizes, int n_in,
                              void* d_out, int out_size, void* d_ws, size_t ws_size,
                              hipStream_t stream) {
  const int*   numbers = (const int*)  d_in[0];
  const float* adj     = (const float*)d_in[1];
  const float* emb     = (const float*)d_in[2];
  const float* W0      = (const float*)d_in[3];
  const float* a0      = (const float*)d_in[4];
  const float* W1      = (const float*)d_in[5];
  const float* a1      = (const float*)d_in[6];
  const float* W2      = (const float*)d_in[7];
  const float* a2      = (const float*)d_in[8];
  const float* P1w     = (const float*)d_in[9];
  const float* P1b     = (const float*)d_in[10];
  const float* P2w     = (const float*)d_in[11];
  const float* P2b     = (const float*)d_in[12];
  float* out = (float*)d_out;

  const int Bn = in_sizes[0] / NODES;

  // bf16 workspace layout (weights pre-transposed to [N,K] for B fragments)
  __bf16* embb = (__bf16*)d_ws;           // 46*128
  __bf16* Wt0  = embb + 46 * EMB;         // [256,128]
  __bf16* Wt1  = Wt0 + EMB * HID;         // [256,256]
  __bf16* Wt2  = Wt1 + HID * HID;         // [256,256]
  __bf16* P1t  = Wt2 + HID * HID;         // [512,1536]
  __bf16* P2t  = P1t + REPR * PROJ;       // [128,512]

  auto cvt = [&](const float* s, __bf16* d, int R, int C, int tr) {
    const int n = R * C;
    hipLaunchKernelGGL(cvt_bf16_kernel, dim3((n + 255) / 256), dim3(256), 0, stream,
                       s, d, R, C, tr);
  };
  cvt(emb, embb, 46, EMB, 0);
  cvt(W0,  Wt0,  EMB, HID, 1);
  cvt(W1,  Wt1,  HID, HID, 1);
  cvt(W2,  Wt2,  HID, HID, 1);
  cvt(P1w, P1t,  REPR, PROJ, 1);
  cvt(P2w, P2t,  PROJ, 128, 1);

  const int nBlocks = (Bn + SPB - 1) / SPB;
  hipLaunchKernelGGL(gat_vae_fused_kernel, dim3(nBlocks), dim3(256), 0, stream,
                     numbers, adj, embb, Wt0, Wt1, Wt2, a0, a1, a2,
                     P1t, P1b, P2t, P2b, out, Bn);
}
